// TransConvEncoderModule_39367670235534
// MI455X (gfx1250) — matrix operand order repaired and checked by reading, not compiled, not run on hardware
//
#include <hip/hip_runtime.h>
#include <hip/hip_bf16.h>

// ---------------------------------------------------------------------------
// CDNA5 (gfx1250) bf16 WMMA implementation of the TransConv attention encoder.
// Each wave computes a 32x64 output tile (8 accumulators, 2 A-frags x 4 B-
// frags per 32-K slice), with a 2-deep software pipeline (load next slice
// while WMMA-ing the current one).  Conv inputs are spatially zero-padded so
// the inner loop is completely branch-free (no EXEC divergence around loads).
// ---------------------------------------------------------------------------

typedef __attribute__((ext_vector_type(16))) __bf16 v16bf;
typedef __attribute__((ext_vector_type(8)))  float  v8f;

union Frag16 { v16bf v; int4 q[2]; };

// ---------------------------------------------------------------------------
// Generic bf16 GEMM / implicit-conv kernel.
//   Out[m][n] = act( (sum_k A[m][k] * Bt[n][k]) * scale[n] + shift[n] + pos )
// A row-major [Mpad][lda] bf16; for ntaps==9 A is a spatially padded NHWC
// image ((convH+2) x (convW+2) rows per batch image) so all tap loads are
// valid.  Bt row-major [Npad][ldb] (n-major, K-contiguous), 9 taps at
// tapStride for convs.  Requires Mpad%32==0, Npad%64==0, K%64==0.
// storeMode: 0 = [m][n] (batch-split), 1 = transposed [n][m], 2 = padded NHWC.
// ---------------------------------------------------------------------------
__global__ __launch_bounds__(256) void wmma_gemm_bf16(
    const __hip_bfloat16* __restrict__ A,
    const __hip_bfloat16* __restrict__ Bt,
    void* __restrict__ Out,
    int Mpad, int Npad, int K, int mValid, int nValid,
    int lda, int ldb, int ldo,
    long long sA, long long sB, long long sO,        // per-batch strides (grid.y)
    int ntaps, int convH, int convW, long long tapStride,
    const float* __restrict__ scaleVec,
    const float* __restrict__ shiftVec,
    const float* __restrict__ posAdd, int posMod, int posLd,
    int reluFlag, int outF32,
    int mSplit, long long splitStride, int storeMode)
{
    const int lane    = threadIdx.x & 31;
    const int halfsel = lane >> 4;       // 0: lanes 0-15, 1: lanes 16-31
    const int lane16  = lane & 15;

    const int mTiles  = Mpad >> 5;       // 32-row tiles
    const int nGroups = Npad >> 6;       // 64-col groups

    const int wave = blockIdx.x * (blockDim.x >> 5) + (threadIdx.x >> 5);
    if (wave >= mTiles * nGroups) return;

    const int mTile  = wave / nGroups;
    const int nGroup = wave - mTile * nGroups;
    const int m0 = mTile << 5;
    const int n0 = nGroup << 6;

    const int batch = blockIdx.y;
    const __hip_bfloat16* Ab = A  + (size_t)batch * (size_t)sA;
    const __hip_bfloat16* Bb = Bt + (size_t)batch * (size_t)sB;

    // Per-lane A rows (WMMA A layout: lane16 selects the M row).
    const int mr0 = m0 + lane16;
    const int mr1 = mr0 + 16;
    long long arow0, arow1;
    const int Wp = convW + 2;
    if (ntaps > 1) {
        const int pixPerB = convH * convW;
        const int padHW   = (convH + 2) * Wp;
        const int b0 = mr0 / pixPerB, p0 = mr0 - b0 * pixPerB;
        const int b1 = mr1 / pixPerB, p1 = mr1 - b1 * pixPerB;
        const int h0 = p0 / convW,    w0 = p0 - h0 * convW;
        const int h1 = p1 / convW,    w1 = p1 - h1 * convW;
        arow0 = (long long)b0 * padHW + (h0 + 1) * Wp + (w0 + 1);
        arow1 = (long long)b1 * padHW + (h1 + 1) * Wp + (w1 + 1);
    } else {
        arow0 = mr0;
        arow1 = mr1;
    }

    v8f acc00 = {0.f, 0.f, 0.f, 0.f, 0.f, 0.f, 0.f, 0.f};
    v8f acc01 = acc00, acc02 = acc00, acc03 = acc00;
    v8f acc10 = acc00, acc11 = acc00, acc12 = acc00, acc13 = acc00;

    const int offA  = halfsel << 4;      // A frag half offset (elements)
    const int offBh = halfsel << 4;      // B frag half offset (elements)

// Fragment variables are passed explicitly (no token pasting against '.').
#define ATH_LOAD(FA0, FA1, FB0, FB1, FB2, FB3, KOFF)                             \
    {                                                                            \
        const int kk_ = (KOFF);                                                  \
        const __hip_bfloat16* p_ = aRow0 + kk_ + offA;                           \
        FA0.q[0] = *(const int4*)(p_);                                           \
        FA0.q[1] = *(const int4*)(p_ + 8);                                       \
        p_ = aRow1 + kk_ + offA;                                                 \
        FA1.q[0] = *(const int4*)(p_);                                           \
        FA1.q[1] = *(const int4*)(p_ + 8);                                       \
        const __hip_bfloat16* q_ = Brow + kk_ + offBh;                           \
        FB0.q[0] = *(const int4*)(q_);                                           \
        FB0.q[1] = *(const int4*)(q_ + 8);  q_ += (size_t)16 * ldb;              \
        FB1.q[0] = *(const int4*)(q_);                                           \
        FB1.q[1] = *(const int4*)(q_ + 8);  q_ += (size_t)16 * ldb;              \
        FB2.q[0] = *(const int4*)(q_);                                           \
        FB2.q[1] = *(const int4*)(q_ + 8);  q_ += (size_t)16 * ldb;              \
        FB3.q[0] = *(const int4*)(q_);                                           \
        FB3.q[1] = *(const int4*)(q_ + 8);                                       \
    }

#define ATH_WMMA(FA0, FA1, FB0, FB1, FB2, FB3)                                   \
    {                                                                            \
        acc00 = __builtin_amdgcn_wmma_f32_16x16x32_bf16(                         \
            false, FA0.v, false, FB0.v, (short)0, acc00, false, false);          \
        acc01 = __builtin_amdgcn_wmma_f32_16x16x32_bf16(                         \
            false, FA0.v, false, FB1.v, (short)0, acc01, false, false);          \
        acc02 = __builtin_amdgcn_wmma_f32_16x16x32_bf16(                         \
            false, FA0.v, false, FB2.v, (short)0, acc02, false, false);          \
        acc03 = __builtin_amdgcn_wmma_f32_16x16x32_bf16(                         \
            false, FA0.v, false, FB3.v, (short)0, acc03, false, false);          \
        acc10 = __builtin_amdgcn_wmma_f32_16x16x32_bf16(                         \
            false, FA1.v, false, FB0.v, (short)0, acc10, false, false);          \
        acc11 = __builtin_amdgcn_wmma_f32_16x16x32_bf16(                         \
            false, FA1.v, false, FB1.v, (short)0, acc11, false, false);          \
        acc12 = __builtin_amdgcn_wmma_f32_16x16x32_bf16(                         \
            false, FA1.v, false, FB2.v, (short)0, acc12, false, false);          \
        acc13 = __builtin_amdgcn_wmma_f32_16x16x32_bf16(                         \
            false, FA1.v, false, FB3.v, (short)0, acc13, false, false);          \
    }

    for (int tap = 0; tap < ntaps; ++tap) {
        long long tapOff = 0;
        if (ntaps > 1) {
            const int dy = tap / 3 - 1;
            const int dx = tap - (tap / 3) * 3 - 1;
            tapOff = (long long)dy * Wp + dx;
        }
        const __hip_bfloat16* aRow0 = Ab + (arow0 + tapOff) * (size_t)lda;
        const __hip_bfloat16* aRow1 = Ab + (arow1 + tapOff) * (size_t)lda;
        const __hip_bfloat16* Brow  = Bb + (size_t)tap * (size_t)tapStride
                                        + (size_t)(n0 + lane16) * (size_t)ldb;

        Frag16 aA0, aA1, bA0, bA1, bA2, bA3;
        Frag16 aB0, aB1, bB0, bB1, bB2, bB3;

        ATH_LOAD(aA0, aA1, bA0, bA1, bA2, bA3, 0)
        for (int k0 = 0; k0 < K; k0 += 64) {
            ATH_LOAD(aB0, aB1, bB0, bB1, bB2, bB3, k0 + 32)  // K%64==0 -> valid
            if (k0 + 128 < K)
                __builtin_prefetch((const void*)(aRow0 + k0 + 128), 0, 1);
            ATH_WMMA(aA0, aA1, bA0, bA1, bA2, bA3)
            if (k0 + 64 < K) {
                ATH_LOAD(aA0, aA1, bA0, bA1, bA2, bA3, k0 + 64)
            }
            ATH_WMMA(aB0, aB1, bB0, bB1, bB2, bB3)
        }
    }
#undef ATH_LOAD
#undef ATH_WMMA

    // Epilogue.  D layout: elem v -> M = tile_m + v + 8*half, N = n-lane.
    const long long ob = (long long)batch * sO;
    auto storeTile = [&](const v8f& accv, int mi, int j) {
        const int n = n0 + (j << 4) + lane16;
        if (n >= nValid) return;
        const float sc = scaleVec ? scaleVec[n] : 1.0f;
        const float sh = shiftVec ? shiftVec[n] : 0.0f;
#pragma unroll
        for (int v = 0; v < 8; ++v) {
            const int m = m0 + (mi << 4) + v + (halfsel << 3);
            if (m >= mValid) continue;
            float val = accv[v] * sc + sh;
            if (posAdd) val += posAdd[(size_t)(m % posMod) * (size_t)posLd + n];
            if (reluFlag) val = fmaxf(val, 0.0f);
            const long long mq = m / mSplit;
            const long long mr = m - mq * (long long)mSplit;
            long long idx;
            if (storeMode == 1) {
                idx = ob + mq * splitStride + (long long)n * ldo + mr;
            } else if (storeMode == 2) {               // padded NHWC (conv geom)
                const int hh = (int)(mr / convW);
                const int ww = (int)(mr - (long long)hh * convW);
                idx = ob + mq * splitStride
                    + (long long)((hh + 1) * Wp + (ww + 1)) * ldo + n;
            } else {
                idx = ob + mq * splitStride + mr * (long long)ldo + n;
            }
            if (outF32) ((float*)Out)[idx] = val;
            else ((__hip_bfloat16*)Out)[idx] = __float2bfloat16(val);
        }
    };
    storeTile(acc00, 0, 0); storeTile(acc01, 0, 1);
    storeTile(acc02, 0, 2); storeTile(acc03, 0, 3);
    storeTile(acc10, 1, 0); storeTile(acc11, 1, 1);
    storeTile(acc12, 1, 2); storeTile(acc13, 1, 3);
}

// ----------------------------- helper kernels ------------------------------

// NCHW fp32 -> spatially padded NHWC bf16 ((H+2)x(W+2) rows per image).
__global__ void k_nchw_to_pad_bf16(const float* __restrict__ src,
                                   __hip_bfloat16* __restrict__ dst,
                                   int Bn, int CI, int H, int W)
{
    long long i = (long long)blockIdx.x * blockDim.x + threadIdx.x;
    const long long total = (long long)Bn * H * W * CI;
    if (i >= total) return;
    const long long hwci = (long long)H * W * CI;
    const int b = (int)(i / hwci);
    const long long r = i - (long long)b * hwci;
    const int p = (int)(r / CI);
    const int c = (int)(r - (long long)p * CI);
    const int h = p / W;
    const int w = p - h * W;
    const int Wp = W + 2;
    const long long row = (long long)b * ((H + 2) * Wp) + (h + 1) * Wp + (w + 1);
    dst[row * CI + c] = __float2bfloat16(src[((long long)b * CI + c) * (H * W) + p]);
}

__global__ void k_pack3x3_bf16(const float* __restrict__ w,
                               __hip_bfloat16* __restrict__ wt, int CO, int CI)
{
    long long i = (long long)blockIdx.x * blockDim.x + threadIdx.x;
    const long long total = 9LL * CO * CI;
    if (i >= total) return;
    const long long coci = (long long)CO * CI;
    const int t  = (int)(i / coci);
    const long long r = i - (long long)t * coci;
    const int co = (int)(r / CI);
    const int ci = (int)(r - (long long)co * CI);
    wt[i] = __float2bfloat16(w[((long long)co * CI + ci) * 9 + t]);  // OIHW -> [t][co][ci]
}

__global__ void k_f32_to_bf16(const float* __restrict__ a,
                              __hip_bfloat16* __restrict__ o, long long n)
{
    long long i = (long long)blockIdx.x * blockDim.x + threadIdx.x;
    if (i < n) o[i] = __float2bfloat16(a[i]);
}

__global__ void k_bn_prep(const float* __restrict__ g, const float* __restrict__ beta,
                          const float* __restrict__ mean, const float* __restrict__ var,
                          const float* __restrict__ bias,
                          float* __restrict__ scale, float* __restrict__ shift, int C)
{
    const int i = threadIdx.x + blockIdx.x * blockDim.x;
    if (i >= C) return;
    const float s = g[i] * rsqrtf(var[i] + 1e-5f);
    scale[i] = s;
    shift[i] = (bias[i] - mean[i]) * s + beta[i];
}

__global__ void k_pos_embed(float* __restrict__ posT, int H, int W, int C)
{
    long long i = (long long)blockIdx.x * blockDim.x + threadIdx.x;
    const long long total = (long long)H * W * C;
    if (i >= total) return;
    const int p = (int)(i / C);
    const int c = (int)(i - (long long)p * C);
    const int h = p / W;
    const int w = p - h * W;
    const int F = C >> 1;
    const int cc = (c < F) ? c : (c - F);
    const float coord = (c < F) ? (float)(h + 1) : (float)(w + 1);
    const float ex = (float)(cc & ~1) / (float)F;
    const float v = coord / powf(10000.0f, ex);
    posT[i] = (cc & 1) ? cosf(v) : sinf(v);
}

__global__ __launch_bounds__(256) void k_softmax_row(
    const float* __restrict__ E, __hip_bfloat16* __restrict__ Attn,
    int cols, int ldE, int ldA, long long bsE, long long bsA, int rowsPerB)
{
    __shared__ float red[256];
    const int bid = blockIdx.x;
    const int b = bid / rowsPerB;
    const int j = bid - b * rowsPerB;
    const float* row = E + (long long)b * bsE + (long long)j * ldE;
    __hip_bfloat16* arow = Attn + (long long)b * bsA + (long long)j * ldA;
    const int tid = threadIdx.x;

    float mx = -3.0e38f;
    for (int i = tid; i < cols; i += 256) mx = fmaxf(mx, row[i]);
    red[tid] = mx; __syncthreads();
    for (int s = 128; s > 0; s >>= 1) {
        if (tid < s) red[tid] = fmaxf(red[tid], red[tid + s]);
        __syncthreads();
    }
    mx = red[0]; __syncthreads();

    float sum = 0.f;
    for (int i = tid; i < cols; i += 256) sum += __expf(row[i] - mx);
    red[tid] = sum; __syncthreads();
    for (int s = 128; s > 0; s >>= 1) {
        if (tid < s) red[tid] += red[tid + s];
        __syncthreads();
    }
    const float inv = 1.0f / red[0];

    for (int i = tid; i < ldA; i += 256) {
        const float v = (i < cols) ? __expf(row[i] - mx) * inv : 0.0f;
        arow[i] = __float2bfloat16(v);
    }
}

// y_pad[b, h+1, w+1, c] = gamma * oCM[b, c, p] + x1[(b,p), c]
__global__ void k_residual_pad(const __hip_bfloat16* __restrict__ ocm,
                               const __hip_bfloat16* __restrict__ x1,
                               const float* __restrict__ gamma,
                               __hip_bfloat16* __restrict__ ypad,
                               int C, int H, int W, int ldO, long long bsO,
                               long long total)
{
    long long i = (long long)blockIdx.x * blockDim.x + threadIdx.x;
    if (i >= total) return;
    const int HW = H * W;
    const long long m = i / C;
    const int c = (int)(i - m * C);
    const long long b = m / HW;
    const int p = (int)(m - b * HW);
    const int h = p / W;
    const int w = p - h * W;
    const int Wp = W + 2;
    const float o = __bfloat162float(ocm[b * bsO + (long long)c * ldO + p]);
    const float val = gamma[0] * o + __bfloat162float(x1[i]);
    const long long row = b * ((H + 2) * Wp) + (h + 1) * Wp + (w + 1);
    ypad[row * C + c] = __float2bfloat16(val);
}

__global__ void k_nhwc_to_nchw_f32(const __hip_bfloat16* __restrict__ x,
                                   float* __restrict__ out, int Bn, int C, int HW)
{
    long long i = (long long)blockIdx.x * blockDim.x + threadIdx.x;
    const long long total = (long long)Bn * C * HW;
    if (i >= total) return;
    const int chw = C * HW;
    const int b = (int)(i / chw);
    const int r = (int)(i - (long long)b * chw);
    const int c = r / HW;
    const int p = r - c * HW;
    out[i] = __bfloat162float(x[((long long)b * HW + p) * C + c]);
}

// ------------------------------- host side ---------------------------------

static void launch_gemm(hipStream_t stream,
    const __hip_bfloat16* A, const __hip_bfloat16* Bt, void* Out,
    int Mpad, int Npad, int K, int mValid, int nValid,
    int lda, int ldb, int ldo,
    long long sA, long long sB, long long sO, int batch,
    int ntaps, int convH, int convW, long long tapStride,
    const float* scale, const float* shift,
    const float* posAdd, int posMod, int posLd,
    int relu, int outF32, int mSplit, long long splitStride, int storeMode)
{
    const int waves = (Mpad / 32) * (Npad / 64);
    const int blocks = (waves + 7) / 8;
    dim3 grid((unsigned)blocks, (unsigned)batch);
    wmma_gemm_bf16<<<grid, dim3(256), 0, stream>>>(
        A, Bt, Out, Mpad, Npad, K, mValid, nValid, lda, ldb, ldo,
        sA, sB, sO, ntaps, convH, convW, tapStride,
        scale, shift, posAdd, posMod, posLd,
        relu, outF32, mSplit, splitStride, storeMode);
}

extern "C" void kernel_launch(void* const* d_in, const int* in_sizes, int n_in,
                              void* d_out, int out_size, void* d_ws, size_t ws_size,
                              hipStream_t stream)
{
    (void)in_sizes; (void)n_in; (void)out_size; (void)ws_size;
    const int Bn = 8, H = 20, Wd = 50, HW = H * Wd, M8 = Bn * HW;  // 8000 pixels
    const int C = 256, Rr = 64;
    const int P = 1024;                   // attention dim padding (M,N,K tiles)
    const int Hp = H + 2, Wp = Wd + 2, padHW = Hp * Wp;   // 22*52 = 1144
    const int BIG = 1 << 30;

    const float* src = (const float*)d_in[0];

    char* wp = (char*)d_ws;
    auto alloc = [&](size_t bytes) -> void* {
        void* p = (void*)wp;
        wp += (bytes + 255) & ~(size_t)255;
        return p;
    };
    __hip_bfloat16* xpadA = (__hip_bfloat16*)alloc((size_t)Bn * padHW * 2048 * 2);
    __hip_bfloat16* xpadB = (__hip_bfloat16*)alloc((size_t)Bn * padHW * C * 2);
    __hip_bfloat16* ypadB = (__hip_bfloat16*)alloc((size_t)Bn * padHW * C * 2);
    __hip_bfloat16* x1    = (__hip_bfloat16*)alloc((size_t)M8 * C * 2);
    __hip_bfloat16* xout  = (__hip_bfloat16*)alloc((size_t)M8 * C * 2);
    __hip_bfloat16* wpreT = (__hip_bfloat16*)alloc((size_t)9 * C * 2048 * 2);
    __hip_bfloat16* wfinT = (__hip_bfloat16*)alloc((size_t)9 * C * C * 2);
    __hip_bfloat16* wqT   = (__hip_bfloat16*)alloc((size_t)Rr * C * 2);
    __hip_bfloat16* wkT   = (__hip_bfloat16*)alloc((size_t)Rr * C * 2);
    __hip_bfloat16* wvT   = (__hip_bfloat16*)alloc((size_t)C * C * 2);
    __hip_bfloat16* qT    = (__hip_bfloat16*)alloc((size_t)Bn * P * Rr * 2);
    __hip_bfloat16* kT    = (__hip_bfloat16*)alloc((size_t)Bn * P * Rr * 2);
    __hip_bfloat16* vCM   = (__hip_bfloat16*)alloc((size_t)Bn * C * P * 2);
    __hip_bfloat16* oCM   = (__hip_bfloat16*)alloc((size_t)Bn * C * P * 2);
    float*          energy= (float*)alloc((size_t)Bn * P * P * 4);
    __hip_bfloat16* attn  = (__hip_bfloat16*)alloc((size_t)Bn * P * P * 2);
    float*          posT  = (float*)alloc((size_t)HW * C * 4);
    float* preScale = (float*)alloc(C * 4);
    float* preShift = (float*)alloc(C * 4);
    float* finScale = (float*)alloc(C * 4);
    float* finShift = (float*)alloc(C * 4);

    // Zero padded activation buffers (halo rows) and v K-padding once per call.
    hipMemsetAsync(xpadA, 0, (size_t)Bn * padHW * 2048 * 2, stream);
    hipMemsetAsync(xpadB, 0, (size_t)Bn * padHW * C * 2, stream);
    hipMemsetAsync(ypadB, 0, (size_t)Bn * padHW * C * 2, stream);
    hipMemsetAsync(vCM,   0, (size_t)Bn * C * P * 2, stream);

    auto gridFor = [](long long n) { return dim3((unsigned)((n + 255) / 256)); };

    k_pos_embed<<<gridFor((long long)HW * C), 256, 0, stream>>>(posT, H, Wd, C);
    k_nchw_to_pad_bf16<<<gridFor((long long)Bn * 2048 * HW), 256, 0, stream>>>(
        src, xpadA, Bn, 2048, H, Wd);

    int curCI = 2048;
    for (int L = 0; L < 2; ++L) {
        const float* const* Pm = (const float* const*)(d_in + 1 + L * 19);
        const float *pre_w = Pm[0], *pre_b = Pm[1], *pre_g = Pm[2], *pre_beta = Pm[3];
        const float *pre_m = Pm[4], *pre_vv = Pm[5];
        const float *q_w = Pm[6], *q_b = Pm[7], *k_w = Pm[8], *k_b = Pm[9];
        const float *v_w = Pm[10], *v_b = Pm[11];
        const float *fin_w = Pm[12], *fin_b = Pm[13], *fin_g = Pm[14], *fin_beta = Pm[15];
        const float *fin_m = Pm[16], *fin_vv = Pm[17], *gamma = Pm[18];

        const __hip_bfloat16* xin = (L == 0) ? xpadA : xpadB;

        // --- weight packing + BN folding ---
        k_pack3x3_bf16<<<gridFor(9LL * C * curCI), 256, 0, stream>>>(pre_w, wpreT, C, curCI);
        k_pack3x3_bf16<<<gridFor(9LL * C * C), 256, 0, stream>>>(fin_w, wfinT, C, C);
        k_f32_to_bf16<<<gridFor((long long)Rr * C), 256, 0, stream>>>(q_w, wqT, (long long)Rr * C);
        k_f32_to_bf16<<<gridFor((long long)Rr * C), 256, 0, stream>>>(k_w, wkT, (long long)Rr * C);
        k_f32_to_bf16<<<gridFor((long long)C * C), 256, 0, stream>>>(v_w, wvT, (long long)C * C);
        k_bn_prep<<<1, C, 0, stream>>>(pre_g, pre_beta, pre_m, pre_vv, pre_b, preScale, preShift, C);
        k_bn_prep<<<1, C, 0, stream>>>(fin_g, fin_beta, fin_m, fin_vv, fin_b, finScale, finShift, C);

        // --- pre conv 3x3 (implicit GEMM on padded input) + BN + pos -> x1 ---
        launch_gemm(stream, xin, wpreT, x1,
                    M8, C, curCI, M8, C, curCI, curCI, C,
                    0, 0, 0, 1,
                    9, H, Wd, (long long)C * curCI,
                    preScale, preShift, posT, HW, C,
                    0, 0, BIG, 0, 0);

        // --- q = relu(1x1 conv), stored [b][1024][64] ---
        launch_gemm(stream, x1, wqT, qT,
                    M8, Rr, C, M8, Rr, C, C, Rr,
                    0, 0, 0, 1,
                    1, 1, 1, 0,
                    nullptr, q_b, nullptr, 1, 1,
                    1, 0, HW, (long long)P * Rr, 0);

        // --- k = 1x1 conv, stored [b][1024][64] ---
        launch_gemm(stream, x1, wkT, kT,
                    M8, Rr, C, M8, Rr, C, C, Rr,
                    0, 0, 0, 1,
                    1, 1, 1, 0,
                    nullptr, k_b, nullptr, 1, 1,
                    0, 0, HW, (long long)P * Rr, 0);

        // --- v = 1x1 conv, channel-major [b][256][1024] (transposed store) ---
        launch_gemm(stream, x1, wvT, vCM,
                    M8, C, C, M8, C, C, C, P,
                    0, 0, 0, 1,
                    1, 1, 1, 0,
                    nullptr, v_b, nullptr, 1, 1,
                    0, 0, HW, (long long)C * P, 1);

        // --- energy[b][n][m] = q[b][n]^T k[b][m]  (fp32, batched) ---
        launch_gemm(stream, qT, kT, energy,
                    P, P, Rr, P, P, Rr, Rr, P,
                    (long long)P * Rr, (long long)P * Rr, (long long)P * P, Bn,
                    1, 1, 1, 0,
                    nullptr, nullptr, nullptr, 1, 1,
                    0, 1, BIG, 0, 0);

        // --- row softmax -> attn bf16 [b][1024][1024], zero K-pad columns ---
        k_softmax_row<<<dim3((unsigned)(Bn * HW)), 256, 0, stream>>>(
            energy, attn, HW, P, P,
            (long long)P * P, (long long)P * P, HW);

        // --- out[b][c][j] = sum_i v[b][c][i] * attn[b][j][i]  (batched) ---
        launch_gemm(stream, vCM, attn, oCM,
                    C, P, P, C, P, P, P, P,
                    (long long)C * P, (long long)P * P, (long long)C * P, Bn,
                    1, 1, 1, 0,
                    nullptr, nullptr, nullptr, 1, 1,
                    0, 0, BIG, 0, 0);

        // --- y = gamma * out + x1, written into padded NHWC layout ---
        k_residual_pad<<<gridFor((long long)M8 * C), 256, 0, stream>>>(
            oCM, x1, gamma, ypadB, C, H, Wd, P, (long long)C * P,
            (long long)M8 * C);

        // --- fin conv 3x3 + BN + relu ---
        if (L == 0) {
            // store into padded layout -> becomes next layer's conv input
            launch_gemm(stream, ypadB, wfinT, xpadB,
                        M8, C, C, M8, C, C, C, C,
                        0, 0, 0, 1,
                        9, H, Wd, (long long)C * C,
                        finScale, finShift, nullptr, 1, 1,
                        1, 0, HW, (long long)padHW * C, 2);
        } else {
            launch_gemm(stream, ypadB, wfinT, xout,
                        M8, C, C, M8, C, C, C, C,
                        0, 0, 0, 1,
                        9, H, Wd, (long long)C * C,
                        finScale, finShift, nullptr, 1, 1,
                        1, 0, BIG, 0, 0);
        }
        curCI = C;
    }

    // --- NHWC bf16 -> NCHW fp32 output ---
    k_nhwc_to_nchw_f32<<<gridFor((long long)Bn * C * HW), 256, 0, stream>>>(
        xout, (float*)d_out, Bn, C, HW);
}
